// GraphConv_57251914056097
// MI455X (gfx1250) — matrix operand order repaired and checked by reading, not compiled, not run on hardware
//
#include <hip/hip_runtime.h>
#include <math.h>

// ---------------------------------------------------------------------------
// EdgeConv-style graph conv for MI455X (gfx1250, wave32).
//   y[e,:] = [x_s | x_t - x_s] @ W^T  ==  x_s @ (W1-W2)^T + x_t @ W2^T
// Pass 1: WMMA GEMM per 16-edge tile, accumulate BN batch stats.
// Pass 2: recompute y via WMMA (keeps everything L2-resident, avoids a
//         205 MB y spill), apply BN affine + ELU, scatter-add with
//         global_atomic_add_f32 into channel-major output.
// ---------------------------------------------------------------------------

typedef __attribute__((ext_vector_type(16))) __bf16 v16bf;
typedef __attribute__((ext_vector_type(8)))  float  v8f;

#define IN_C  32
#define TWO_C 64

__device__ __forceinline__ v8f wmma_bf16(v16bf a, v16bf b, v8f c) {
  // (neg_a, A, neg_b, B, c_mod, C, reuse_a, reuse_b)
  return __builtin_amdgcn_wmma_f32_16x16x32_bf16(false, a, false, b, (short)0, c,
                                                 false, false);
}

// B fragments (32x16, K x N), built once per wave.
// Layout: lane L = n + 16*h holds column N=n; elements 0..7 = K[8h..8h+7],
// elements 8..15 = K[16+8h..23+8h] (matches A).
__device__ __forceinline__ void load_B(const float* __restrict__ W, int lane,
                                       v16bf& bWa0, v16bf& bWb0,
                                       v16bf& bWa1, v16bf& bWb1) {
  int n = lane & 15, h = lane >> 4;
#pragma unroll
  for (int j = 0; j < 16; ++j) {
    int k = (j < 8) ? (8 * h + j) : (16 + 8 * h + (j - 8));
    // N-half 0: out channel o = n
    float w1 = W[n * TWO_C + k];
    float w2 = W[n * TWO_C + 32 + k];
    bWa0[j] = (__bf16)(w1 - w2);
    bWb0[j] = (__bf16)w2;
    // N-half 1: out channel o = 16 + n
    float w1b = W[(16 + n) * TWO_C + k];
    float w2b = W[(16 + n) * TWO_C + 32 + k];
    bWa1[j] = (__bf16)(w1b - w2b);
    bWb1[j] = (__bf16)w2b;
  }
}

// A fragment (16x32, M x K), 16-bit layout per ISA 7.12.2:
// lane = m + 16*h holds row m; elements 0..7 = K[8h..8h+7],
// elements 8..15 = K[16+8h..23+8h]. Contiguous f32 reads -> bf16.
__device__ __forceinline__ v16bf load_A_row(const float* __restrict__ row, int h) {
  v16bf a;
#pragma unroll
  for (int j = 0; j < 8; ++j) a[j]     = (__bf16)row[8 * h + j];
#pragma unroll
  for (int j = 0; j < 8; ++j) a[8 + j] = (__bf16)row[16 + 8 * h + j];
  return a;
}

// ---------------------------------------------------------------------------

__global__ void zero_k(float* __restrict__ out, int n, float* __restrict__ stats) {
  int i = blockIdx.x * blockDim.x + threadIdx.x;
  if (i < n) out[i] = 0.0f;
  if (i < 64) stats[i] = 0.0f;  // sum[32], sumsq[32]
}

// Pass 1: per-channel sum / sum-of-squares over all E edges.
__global__ void pass1_wmma(const float* __restrict__ x, const int* __restrict__ src,
                           const int* __restrict__ tgt, const float* __restrict__ W,
                           float* __restrict__ stats, int nTiles) {
  int lane   = threadIdx.x & 31;
  int wave   = (blockIdx.x * blockDim.x + threadIdx.x) >> 5;
  int nWaves = (gridDim.x * blockDim.x) >> 5;
  int n = lane & 15, h = lane >> 4;

  v16bf bWa0, bWb0, bWa1, bWb1;
  load_B(W, lane, bWa0, bWb0, bWa1, bWb1);

  float s0 = 0.f, q0 = 0.f, s1 = 0.f, q1 = 0.f;

  for (int tile = wave; tile < nTiles; tile += nWaves) {
    int e = tile * 16 + n;
    __builtin_prefetch(src + (tile + nWaves) * 16, 0, 1);  // global_prefetch_b8
    int si = src[e], ti = tgt[e];
    v16bf aS = load_A_row(x + si * IN_C, h);
    v16bf aT = load_A_row(x + ti * IN_C, h);

    v8f c0 = {};
    v8f c1 = {};
    c0 = wmma_bf16(aS, bWa0, c0);
    c0 = wmma_bf16(aT, bWb0, c0);
    c1 = wmma_bf16(aS, bWa1, c1);
    c1 = wmma_bf16(aT, bWb1, c1);

#pragma unroll
    for (int m = 0; m < 8; ++m) {  // rows M = m + 8h, channel N = n (+16 for c1)
      float y0 = c0[m], y1 = c1[m];
      s0 += y0; q0 += y0 * y0;
      s1 += y1; q1 += y1 * y1;
    }
  }
  atomicAdd(&stats[n],      s0);
  atomicAdd(&stats[16 + n], s1);
  atomicAdd(&stats[32 + n], q0);
  atomicAdd(&stats[48 + n], q1);
}

// Scalar tail for E % 16 edges (empty for E = 1.6M).
__global__ void pass1_tail(const float* __restrict__ x, const int* __restrict__ src,
                           const int* __restrict__ tgt, const float* __restrict__ W,
                           float* __restrict__ stats, int e0, int E) {
  int e = e0 + blockIdx.x * blockDim.x + threadIdx.x;
  if (e >= E) return;
  int si = src[e], ti = tgt[e];
  const float* xs = x + si * IN_C;
  const float* xt = x + ti * IN_C;
  for (int o = 0; o < 32; ++o) {
    float acc = 0.f;
    for (int c = 0; c < 32; ++c)
      acc += xs[c] * W[o * TWO_C + c] + (xt[c] - xs[c]) * W[o * TWO_C + 32 + c];
    atomicAdd(&stats[o], acc);
    atomicAdd(&stats[32 + o], acc * acc);
  }
}

// Fold batch stats into BN affine: scale = gamma*rsqrt(var+eps), shift = beta - mean*scale
__global__ void bn_params(const float* __restrict__ stats, const float* __restrict__ gamma,
                          const float* __restrict__ beta, float* __restrict__ scsh,
                          float invE) {
  int c = threadIdx.x;
  if (c < 32) {
    float mean = stats[c] * invE;
    float var  = stats[32 + c] * invE - mean * mean;
    float s    = gamma[c] * rsqrtf(var + 1e-5f);
    scsh[c]      = s;
    scsh[32 + c] = beta[c] - mean * s;
  }
}

// Pass 2: recompute y, BN affine + ELU, scatter-add into out[c*Nnodes + src].
__global__ void pass2_wmma(const float* __restrict__ x, const int* __restrict__ src,
                           const int* __restrict__ tgt, const float* __restrict__ W,
                           const float* __restrict__ scsh, float* __restrict__ out,
                           int nTiles, int Nnodes) {
  int lane   = threadIdx.x & 31;
  int wave   = (blockIdx.x * blockDim.x + threadIdx.x) >> 5;
  int nWaves = (gridDim.x * blockDim.x) >> 5;
  int n = lane & 15, h = lane >> 4;

  v16bf bWa0, bWb0, bWa1, bWb1;
  load_B(W, lane, bWa0, bWb0, bWa1, bWb1);

  float sc0 = scsh[n],      sh0 = scsh[32 + n];
  float sc1 = scsh[16 + n], sh1 = scsh[48 + n];

  for (int tile = wave; tile < nTiles; tile += nWaves) {
    int e = tile * 16 + n;
    __builtin_prefetch(src + (tile + nWaves) * 16, 0, 1);
    int si = src[e], ti = tgt[e];
    v16bf aS = load_A_row(x + si * IN_C, h);
    v16bf aT = load_A_row(x + ti * IN_C, h);

    v8f c0 = {};
    v8f c1 = {};
    c0 = wmma_bf16(aS, bWa0, c0);
    c0 = wmma_bf16(aT, bWb0, c0);
    c1 = wmma_bf16(aS, bWa1, c1);
    c1 = wmma_bf16(aT, bWb1, c1);

#pragma unroll
    for (int m = 0; m < 8; ++m) {
      int srow = src[tile * 16 + m + 8 * h];  // row M = m + 8h of this tile
      float y0 = c0[m] * sc0 + sh0;
      y0 = y0 > 0.f ? y0 : expm1f(y0);        // ELU(alpha=1)
      atomicAdd(&out[n * Nnodes + srow], y0);
      float y1 = c1[m] * sc1 + sh1;
      y1 = y1 > 0.f ? y1 : expm1f(y1);
      atomicAdd(&out[(16 + n) * Nnodes + srow], y1);
    }
  }
}

__global__ void pass2_tail(const float* __restrict__ x, const int* __restrict__ src,
                           const int* __restrict__ tgt, const float* __restrict__ W,
                           const float* __restrict__ scsh, float* __restrict__ out,
                           int e0, int E, int Nnodes) {
  int e = e0 + blockIdx.x * blockDim.x + threadIdx.x;
  if (e >= E) return;
  int si = src[e], ti = tgt[e];
  const float* xs = x + si * IN_C;
  const float* xt = x + ti * IN_C;
  for (int o = 0; o < 32; ++o) {
    float acc = 0.f;
    for (int c = 0; c < 32; ++c)
      acc += xs[c] * W[o * TWO_C + c] + (xt[c] - xs[c]) * W[o * TWO_C + 32 + c];
    float y = acc * scsh[o] + scsh[32 + o];
    y = y > 0.f ? y : expm1f(y);
    atomicAdd(&out[o * Nnodes + si], y);
  }
}

// ---------------------------------------------------------------------------

extern "C" void kernel_launch(void* const* d_in, const int* in_sizes, int n_in,
                              void* d_out, int out_size, void* d_ws, size_t ws_size,
                              hipStream_t stream) {
  const float* x     = (const float*)d_in[0];   // [N, 32] row-major (B=1)
  const int*   ei    = (const int*)  d_in[1];   // [2, E] int32
  const float* W     = (const float*)d_in[2];   // [32, 64]
  const float* gamma = (const float*)d_in[3];   // [32]
  const float* beta  = (const float*)d_in[4];   // [32]
  float* out = (float*)d_out;                   // [32, N] channel-major
  float* ws  = (float*)d_ws;

  const int E      = in_sizes[1] / 2;
  const int Nnodes = in_sizes[0] / IN_C;
  const int* src = ei;
  const int* tgt = ei + E;
  float* stats = ws;        // sum[32], sumsq[32]
  float* scsh  = ws + 64;   // scale[32], shift[32]

  zero_k<<<(out_size + 255) / 256, 256, 0, stream>>>(out, out_size, stats);

  const int nTiles = E / 16;
  const int rem    = E - nTiles * 16;
  const int blocks = 2048, tpb = 256;  // 16K waves, ~6 tiles/wave

  if (nTiles > 0)
    pass1_wmma<<<blocks, tpb, 0, stream>>>(x, src, tgt, W, stats, nTiles);
  if (rem > 0)
    pass1_tail<<<1, 32, 0, stream>>>(x, src, tgt, W, stats, nTiles * 16, E);

  bn_params<<<1, 32, 0, stream>>>(stats, gamma, beta, scsh, 1.0f / (float)E);

  if (nTiles > 0)
    pass2_wmma<<<blocks, tpb, 0, stream>>>(x, src, tgt, W, scsh, out, nTiles, Nnodes);
  if (rem > 0)
    pass2_tail<<<1, 32, 0, stream>>>(x, src, tgt, W, scsh, out, nTiles * 16, E, Nnodes);
}
